// _LSTMCell_6803228196930
// MI455X (gfx1250) — 8-bit matrix operand order repaired and checked by reading, compiled, not run on hardware
//
#include <hip/hip_runtime.h>
#include <math.h>

typedef __attribute__((ext_vector_type(8))) int v8i;
typedef __attribute__((ext_vector_type(4))) int v4i;
typedef __attribute__((ext_vector_type(2))) int v2i;

#define B_ROWS 8192
#define DDIM   1024
#define HDIM   1024
#define FH     4096

#define ROWS_PER_WG 128
#define LDS_PITCH   1040
#define LDS_BYTES   (64 * LDS_PITCH)

__device__ __forceinline__ unsigned enc_f(float f) {
    unsigned u = __float_as_uint(f);
    return (u & 0x80000000u) ? ~u : (u | 0x80000000u);
}
__device__ __forceinline__ float dec_f(unsigned e) {
    unsigned u = (e & 0x80000000u) ? (e ^ 0x80000000u) : ~e;
    return __uint_as_float(u);
}

__global__ void mm_init_kernel(unsigned* mm) {
    int t = threadIdx.x;
    if (t < 8) mm[t] = (t & 1) ? 0u : 0xFFFFFFFFu;
}

__global__ void minmax_kernel(const float* __restrict__ x, int n, unsigned* __restrict__ mm) {
    __shared__ unsigned smin[256], smax[256];
    unsigned lmin = 0xFFFFFFFFu, lmax = 0u;
    for (int i = blockIdx.x * blockDim.x + threadIdx.x; i < n; i += gridDim.x * blockDim.x) {
        unsigned e = enc_f(x[i]);
        lmin = min(lmin, e);
        lmax = max(lmax, e);
    }
    smin[threadIdx.x] = lmin; smax[threadIdx.x] = lmax;
    __syncthreads();
    for (int s = 128; s > 0; s >>= 1) {
        if ((int)threadIdx.x < s) {
            smin[threadIdx.x] = min(smin[threadIdx.x], smin[threadIdx.x + s]);
            smax[threadIdx.x] = max(smax[threadIdx.x], smax[threadIdx.x + s]);
        }
        __syncthreads();
    }
    if (threadIdx.x == 0) {
        atomicMin(&mm[0], smin[0]);
        atomicMax(&mm[1], smax[0]);
    }
}

__global__ void quant_rows_kernel(const float* __restrict__ src,
                                  unsigned char* __restrict__ q,
                                  float* __restrict__ rowsum,
                                  const unsigned* __restrict__ mm,
                                  const int* __restrict__ nbp) {
    __shared__ int ssum[256];
    const int row = blockIdx.x;
    const int t   = threadIdx.x;
    const int qmax_i = (1 << nbp[0]) - 1;
    const float mn = dec_f(mm[0]);
    const float mx = dec_f(mm[1]);
    const float scale = (mx - mn) / (float)qmax_i;
    const float inv = (scale != 0.f) ? (1.f / scale) : 0.f;

    const float4 v4 = *(const float4*)(src + (size_t)row * DDIM + t * 4);
    const float vals[4] = {v4.x, v4.y, v4.z, v4.w};
    int qq[4];
    #pragma unroll
    for (int j = 0; j < 4; ++j) {
        int v = (int)rintf((vals[j] - mn) * inv);
        v = v < 0 ? 0 : (v > qmax_i ? qmax_i : v);
        qq[j] = v;
    }
    uchar4 pack = make_uchar4((unsigned char)qq[0], (unsigned char)qq[1],
                              (unsigned char)qq[2], (unsigned char)qq[3]);
    *(uchar4*)(q + (size_t)row * DDIM + t * 4) = pack;

    ssum[t] = qq[0] + qq[1] + qq[2] + qq[3];
    __syncthreads();
    for (int s2 = 128; s2 > 0; s2 >>= 1) {
        if (t < s2) ssum[t] += ssum[t + s2];
        __syncthreads();
    }
    if (t == 0) rowsum[row] = (float)ssum[0];
}

__global__ __launch_bounds__(256)
void lstm_wmma_kernel(const unsigned char* __restrict__ qX,
                      const unsigned char* __restrict__ qH,
                      const unsigned char* __restrict__ qWin,
                      const unsigned char* __restrict__ qWh,
                      const float* __restrict__ Sx,  const float* __restrict__ Sh,
                      const float* __restrict__ Swi, const float* __restrict__ Swh,
                      const unsigned* __restrict__ mm,
                      const float* __restrict__ b_in, const float* __restrict__ b_h,
                      const float* __restrict__ cin,
                      float* __restrict__ hout, float* __restrict__ cout,
                      const int* __restrict__ nbp) {
    extern __shared__ unsigned char smem[];

    const int tid  = threadIdx.x;
    const int wave = tid >> 5;
    const int lane = tid & 31;
    const int m    = lane & 15;
    const int hi   = lane >> 4;
    const int row_base = blockIdx.x * ROWS_PER_WG + wave * 16;
    const int col_base = blockIdx.y * 16;
    const int col = col_base + m;

    const float qmaxv = (float)((1 << nbp[0]) - 1);
    const float mnx = dec_f(mm[0]), sxs = (dec_f(mm[1]) - mnx) / qmaxv;
    const float mnh = dec_f(mm[2]), shs = (dec_f(mm[3]) - mnh) / qmaxv;
    const float mwi = dec_f(mm[4]), swi = (dec_f(mm[5]) - mwi) / qmaxv;
    const float mwh = dec_f(mm[6]), swh = (dec_f(mm[7]) - mwh) / qmaxv;

    float gate[4][8];
    #pragma unroll
    for (int g = 0; g < 4; ++g)
        #pragma unroll
        for (int r = 0; r < 8; ++r) gate[g][r] = 0.f;

    const v8i vzero = {0, 0, 0, 0, 0, 0, 0, 0};

    #pragma unroll
    for (int path = 0; path < 2; ++path) {
        const unsigned char* qA = path ? qH  : qX;
        const unsigned char* qW = path ? qWh : qWin;
        const float* Sa = path ? Sh  : Sx;
        const float* Sw = path ? Swh : Swi;
        const float mna = path ? mnh : mnx;
        const float sa  = path ? shs : sxs;
        const float mnw = path ? mwh : mwi;
        const float sw  = path ? swh : swi;

        __syncthreads();
        {
            const int r  = tid >> 2;
            const int q4 = tid & 3;
            const int g  = r >> 4;
            const int n  = r & 15;
            const unsigned char* src =
                qW + (size_t)(g * HDIM + col_base + n) * DDIM + q4 * 256;
            unsigned char* dst = smem + (size_t)r * LDS_PITCH + q4 * 256;
            #pragma unroll
            for (int i = 0; i < 16; ++i)
                *(v4i*)(dst + i * 16) = *(const v4i*)(src + i * 16);
        }
        __syncthreads();

        v8i acc[4] = {vzero, vzero, vzero, vzero};

        const unsigned char* arow = qA + (size_t)(row_base + m) * DDIM;
        for (int k0 = 0; k0 < DDIM; k0 += 64) {
            v8i A;
            {
                const unsigned char* p = arow + k0 + hi * 16;
                v4i t0 = *(const v4i*)(p);
                v4i t1 = *(const v4i*)(p + 32);
                A[0] = t0[0]; A[1] = t0[1]; A[2] = t0[2]; A[3] = t0[3];
                A[4] = t1[0]; A[5] = t1[1]; A[6] = t1[2]; A[7] = t1[3];
            }
            #pragma unroll
            for (int g = 0; g < 4; ++g) {
                const unsigned char* lw =
                    smem + (size_t)(g * 16 + m) * LDS_PITCH + k0 + hi * 16;
                v4i u0 = *(const v4i*)(lw);
                v4i u1 = *(const v4i*)(lw + 32);
                v8i Bf;
                Bf[0] = u0[0]; Bf[1] = u0[1]; Bf[2] = u0[2]; Bf[3] = u0[3];
                Bf[4] = u1[0]; Bf[5] = u1[1]; Bf[6] = u1[2]; Bf[7] = u1[3];
                acc[g] = __builtin_amdgcn_wmma_i32_16x16x64_iu8(
                    false, A, false, Bf, acc[g], false, false);
            }
        }

        const float kconst = (float)DDIM * mna * mnw;
        #pragma unroll
        for (int g = 0; g < 4; ++g) {
            const float colterm = mna * sw * Sw[g * HDIM + col] + kconst;
            #pragma unroll
            for (int r = 0; r < 8; ++r) {
                const int grow = row_base + r + hi * 8;
                gate[g][r] += sa * sw * (float)acc[g][r] + colterm + mnw * sa * Sa[grow];
            }
        }
    }

    float bi[4], bh[4];
    #pragma unroll
    for (int g = 0; g < 4; ++g) {
        bi[g] = b_in[g * HDIM + col];
        bh[g] = b_h[g * HDIM + col];
    }

    #pragma unroll
    for (int r = 0; r < 8; ++r) {
        const int grow = row_base + r + hi * 8;
        const float gi = gate[0][r] + bi[0] + bh[0];
        const float gf = gate[1][r] + bi[1] + bh[1];
        const float gg = gate[2][r] + bi[2] + bh[2];
        const float go = gate[3][r] + bi[3] + bh[3];
        const float iv = 1.f / (1.f + __expf(-gi));
        const float fv = 1.f / (1.f + __expf(-gf));
        const float gv = tanhf(gg);
        const float ov = 1.f / (1.f + __expf(-go));
        const float cv = cin[(size_t)grow * HDIM + col];
        const float cn = fv * cv + iv * gv;
        const float hn = ov * tanhf(cn);
        hout[(size_t)grow * HDIM + col] = hn;
        cout[(size_t)grow * HDIM + col] = cn;
    }
}

extern "C" void kernel_launch(void* const* d_in, const int* in_sizes, int n_in,
                              void* d_out, int out_size, void* d_ws, size_t ws_size,
                              hipStream_t stream) {
    (void)in_sizes; (void)n_in; (void)out_size; (void)ws_size;
    const float* input = (const float*)d_in[0];
    const float* h     = (const float*)d_in[1];
    const float* c     = (const float*)d_in[2];
    const float* W_in  = (const float*)d_in[3];
    const float* b_in  = (const float*)d_in[4];
    const float* W_h   = (const float*)d_in[5];
    const float* b_h   = (const float*)d_in[6];
    const int*   nb    = (const int*)d_in[7];

    char* ws = (char*)d_ws;
    unsigned* mm = (unsigned*)ws;
    size_t off = 256;
    unsigned char* qX   = (unsigned char*)(ws + off); off += (size_t)B_ROWS * DDIM;
    unsigned char* qH   = (unsigned char*)(ws + off); off += (size_t)B_ROWS * HDIM;
    unsigned char* qWin = (unsigned char*)(ws + off); off += (size_t)FH * DDIM;
    unsigned char* qWh  = (unsigned char*)(ws + off); off += (size_t)FH * HDIM;
    float* Sx  = (float*)(ws + off); off += (size_t)B_ROWS * sizeof(float);
    float* Sh  = (float*)(ws + off); off += (size_t)B_ROWS * sizeof(float);
    float* Swi = (float*)(ws + off); off += (size_t)FH * sizeof(float);
    float* Swh = (float*)(ws + off); off += (size_t)FH * sizeof(float);

    float* hout = (float*)d_out;
    float* cout = hout + (size_t)B_ROWS * HDIM;

    mm_init_kernel<<<1, 8, 0, stream>>>(mm);
    minmax_kernel<<<512, 256, 0, stream>>>(input, B_ROWS * DDIM, mm + 0);
    minmax_kernel<<<512, 256, 0, stream>>>(h,     B_ROWS * HDIM, mm + 2);
    minmax_kernel<<<256, 256, 0, stream>>>(W_in,  FH * DDIM,     mm + 4);
    minmax_kernel<<<256, 256, 0, stream>>>(W_h,   FH * HDIM,     mm + 6);

    quant_rows_kernel<<<B_ROWS, 256, 0, stream>>>(input, qX,   Sx,  mm + 0, nb);
    quant_rows_kernel<<<B_ROWS, 256, 0, stream>>>(h,     qH,   Sh,  mm + 2, nb);
    quant_rows_kernel<<<FH,     256, 0, stream>>>(W_in,  qWin, Swi, mm + 4, nb);
    quant_rows_kernel<<<FH,     256, 0, stream>>>(W_h,   qWh,  Swh, mm + 6, nb);

    dim3 grid(B_ROWS / ROWS_PER_WG, HDIM / 16);
    lstm_wmma_kernel<<<grid, 256, LDS_BYTES, stream>>>(qX, qH, qWin, qWh,
                                                       Sx, Sh, Swi, Swh,
                                                       mm, b_in, b_h, c,
                                                       hout, cout, nb);
}
